// convolutional_embedding_49907519979551
// MI455X (gfx1250) — matrix operand order repaired and checked by reading, not compiled, not run on hardware
//
#include <hip/hip_runtime.h>
#include <hip/hip_bf16.h>
#include <cstdint>

#define Bdim 8
#define Nlen 4096
#define Dch  512
#define Ggrp 32
#define REDc 128
#define HIDc 129
#define Kexp 4
#define TEMPc 34.0f
#define EPSc 1e-5f

typedef __attribute__((ext_vector_type(16))) __bf16 v16bf;
typedef __attribute__((ext_vector_type(8)))  float  v8f;
typedef __attribute__((ext_vector_type(4)))  unsigned int u32x4;
typedef __attribute__((ext_vector_type(8)))  int  i32x8;
typedef __attribute__((ext_vector_type(4)))  int  i32x4;

#if __has_builtin(__builtin_amdgcn_tensor_load_to_lds) && __has_builtin(__builtin_amdgcn_s_wait_tensorcnt)
#define HAVE_TDM 1
#else
#define HAVE_TDM 0
#endif

union FragU { v16bf v; uint4 q[2]; };

__device__ inline v8f vzero8() {
  v8f z;
#pragma unroll
  for (int i = 0; i < 8; ++i) z[i] = 0.f;
  return z;
}

// A-matrix fragment (16x32 bf16, MxK): lane L -> row M=L&15; lanes 0-15 hold
// K=[k0..k0+7] and [k0+16..k0+23]; lanes 16-31 hold K=+8 of each (ISA 7.12.2).
__device__ inline v16bf frag_a(const __bf16* base, int ld, int k0, int lane) {
  const int row = lane & 15, half = lane >> 4;
  const __bf16* p = base + (size_t)row * ld + k0 + half * 8;
  FragU f;
  f.q[0] = *reinterpret_cast<const uint4*>(p);
  f.q[1] = *reinterpret_cast<const uint4*>(p + 16);
  return f.v;
}

// B-matrix fragment (32x16 bf16, KxN) sourced from a row-major [o][d] weight:
// lane L -> column N=o0+(L&15); lanes 0-15 hold K=[k0..k0+7] and [k0+16..k0+23],
// lanes 16-31 hold K=+8 of each (same K order as frag_a).
__device__ inline v16bf frag_b(const __bf16* wrow, int ld, int o0, int k0, int lane) {
  const int col = lane & 15, kh = lane >> 4;
  const __bf16* p = wrow + (size_t)(o0 + col) * ld + k0 + kh * 8;
  FragU f;
  f.q[0] = *reinterpret_cast<const uint4*>(p);
  f.q[1] = *reinterpret_cast<const uint4*>(p + 16);
  return f.v;
}

__device__ inline v8f wmma_bf16(v16bf a, v16bf b, v8f c) {
  return __builtin_amdgcn_wmma_f32_16x16x32_bf16(false, a, false, b, (short)0, c,
                                                 false, false);
}

#if HAVE_TDM
// Issue one 2D TDM descriptor: copy `rows` x 512 bf16 tile (row stride 512)
// from global `gsrc` into LDS at `lds_off`. Wave-level op; caller gates to one
// wave and must barrier before LDS is consumed (the tensorcnt wait is done here).
__device__ inline void tdm_tile_512bf16(const __bf16* gsrc, unsigned lds_off, int rows) {
  const unsigned long long ga = (unsigned long long)(uintptr_t)gsrc;
  u32x4 g0;
  g0[0] = 1u;  // count=1, user mode
  g0[1] = lds_off;
  g0[2] = (unsigned)(ga & 0xFFFFFFFFull);
  g0[3] = (unsigned)((ga >> 32) & 0x1FFFFFFull) | (2u << 30);  // type=2 (image)
  i32x8 g1;
  g1[0] = (1 << 16);            // data_size = 2 bytes
  g1[1] = (int)(Dch << 16);     // tensor_dim0 = 512  (lo16 @ bits 63:48)
  g1[2] = (int)(Nlen << 16);    // tensor_dim1 = 4096 (lo16 @ bits 95:80)
  g1[3] = (int)(Dch << 16);     // tile_dim0 = 512    (bits 127:112)
  g1[4] = rows;                 // tile_dim1          (bits 143:128)
  g1[5] = Dch;                  // tensor_dim0_stride = 512
  g1[6] = 0;
  g1[7] = 0;
  i32x4 g2; g2[0] = 0; g2[1] = 0; g2[2] = 0; g2[3] = 0;
  i32x4 g3 = g2;
  i32x8 g4;
#pragma unroll
  for (int i = 0; i < 8; ++i) g4[i] = 0;
  __builtin_amdgcn_tensor_load_to_lds(g0, g1, g2, g3, g4, 0);
  __builtin_amdgcn_s_wait_tensorcnt(0);
}
#endif

// ---------------------------------------------------------------- converts
__global__ void k_f32_to_bf16(const float* __restrict__ s, __bf16* __restrict__ d, int n) {
  int i = blockIdx.x * blockDim.x + threadIdx.x;
  if (i < n) d[i] = (__bf16)s[i];
}

// ---------------------------------------------------------------- pooling
__global__ void k_pool(const float* __restrict__ x, float* __restrict__ pooled) {
  const int b = blockIdx.x, d = threadIdx.x;
  const float* p = x + (size_t)b * Nlen * Dch + d;
  float s = 0.f;
  for (int n = 0; n < Nlen; ++n) s += p[(size_t)n * Dch];
  pooled[b * Dch + d] = s * (1.0f / Nlen);
}

// -------------------------------------------- CondConv attention + Wb mix
__global__ void __launch_bounds__(256) k_att(
    const float* __restrict__ pooled, const float* __restrict__ fc1_w,
    const float* __restrict__ fc2_w, const float* __restrict__ fc2_b,
    const float* __restrict__ dyn_w, const float* __restrict__ dyn_b,
    __bf16* __restrict__ Wb, float* __restrict__ bbv) {
  __shared__ float aH[HIDc];
  __shared__ float lg[Kexp];
  const int b = blockIdx.x, tid = threadIdx.x;
  const float* pb = pooled + b * Dch;
  if (tid < HIDc) {
    const float* w = fc1_w + (size_t)tid * Dch;
    float s = 0.f;
    for (int i = 0; i < Dch; ++i) s += pb[i] * w[i];
    aH[tid] = fmaxf(s, 0.f);
  }
  __syncthreads();
  if (tid < Kexp) {
    const float* w = fc2_w + tid * HIDc;
    float s = fc2_b[tid];
    for (int i = 0; i < HIDc; ++i) s += aH[i] * w[i];
    lg[tid] = s * (1.0f / TEMPc);
  }
  __syncthreads();
  const float m = fmaxf(fmaxf(lg[0], lg[1]), fmaxf(lg[2], lg[3]));
  float e0 = __expf(lg[0] - m), e1 = __expf(lg[1] - m);
  float e2 = __expf(lg[2] - m), e3 = __expf(lg[3] - m);
  const float inv = 1.f / (e0 + e1 + e2 + e3);
  const float a0 = e0 * inv, a1 = e1 * inv, a2 = e2 * inv, a3 = e3 * inv;
  const size_t DD = (size_t)Dch * Dch;
  for (size_t i = tid; i < DD; i += blockDim.x) {
    float s = a0 * dyn_w[i] + a1 * dyn_w[DD + i] + a2 * dyn_w[2 * DD + i] +
              a3 * dyn_w[3 * DD + i];
    Wb[b * DD + i] = (__bf16)s;
  }
  for (int o = tid; o < Dch; o += blockDim.x)
    bbv[b * Dch + o] = a0 * dyn_b[o] + a1 * dyn_b[Dch + o] +
                       a2 * dyn_b[2 * Dch + o] + a3 * dyn_b[3 * Dch + o];
}

// --------------------------- involution gate: two chained WMMA GEMMs
// block = 128 threads (4 waves), 64 rows; wave handles a 16-row tile.
// x tile staged to LDS by the TDM; k-loop software-pipelined.
__global__ void __launch_bounds__(128) k_gate(
    const __bf16* __restrict__ xb, const __bf16* __restrict__ w1b,
    const float* __restrict__ bn_g, const float* __restrict__ bn_b,
    const float* __restrict__ bn_m, const float* __restrict__ bn_v,
    const __bf16* __restrict__ w2b, const float* __restrict__ conv2_b,
    float* __restrict__ gate) {
  __shared__ __align__(16) __bf16 AsG[64 * Dch];        // 64 KB x tile
  __shared__ __align__(16) __bf16 Hs[4][16][REDc + 8];  // bf16 h, A-layout
  const int tid = threadIdx.x;
  const int w = tid >> 5, lane = tid & 31;
  const int col = lane & 15, half = lane >> 4;
  const int blk0 = blockIdx.x * 64;
  const int row0 = blk0 + w * 16;

  // ---- stage 64 x 512 bf16 tile into LDS ----
  const __bf16* gsrc = xb + (size_t)blk0 * Dch;
#if HAVE_TDM
  if (w == 0) tdm_tile_512bf16(gsrc, (unsigned)(uintptr_t)(&AsG[0]), 64);
#else
  {
    const uint4* s4 = reinterpret_cast<const uint4*>(gsrc);
    uint4* d4 = reinterpret_cast<uint4*>(&AsG[0]);
    for (int i = tid; i < 64 * Dch / 8; i += 128) d4[i] = s4[i];
  }
#endif
  __syncthreads();

  // GEMM1: h[16x128] = x_tile[16x512] x conv1_w^T  (double-buffered k-loop)
  v8f acc[8];
#pragma unroll
  for (int t = 0; t < 8; ++t) acc[t] = vzero8();
  const __bf16* arow = AsG + (size_t)(w * 16) * Dch;
  v16bf af = frag_a(arow, Dch, 0, lane);
  v16bf bf8[8];
#pragma unroll
  for (int t = 0; t < 8; ++t) bf8[t] = frag_b(w1b, Dch, t * 16, 0, lane);
#pragma unroll 1
  for (int kk = 0; kk < 16; ++kk) {
    const int kn = (kk < 15) ? (kk + 1) * 32 : 0;  // next-step fragments
    v16bf afn = frag_a(arow, Dch, kn, lane);
    v16bf bfn[8];
#pragma unroll
    for (int t = 0; t < 8; ++t) bfn[t] = frag_b(w1b, Dch, t * 16, kn, lane);
#pragma unroll
    for (int t = 0; t < 8; ++t) acc[t] = wmma_bf16(af, bf8[t], acc[t]);
    af = afn;
#pragma unroll
    for (int t = 0; t < 8; ++t) bf8[t] = bfn[t];
  }
  // eval-BN + ReLU, write bf16 transposed (C-layout -> A-layout) into LDS
#pragma unroll
  for (int t = 0; t < 8; ++t) {
    const int r = t * 16 + col;
    const float sc = bn_g[r] * rsqrtf(bn_v[r] + EPSc);
    const float bi = bn_b[r] - bn_m[r] * sc;
#pragma unroll
    for (int v = 0; v < 8; ++v) {
      const float h = fmaxf(acc[t][v] * sc + bi, 0.f);
      Hs[w][v + 8 * half][r] = (__bf16)h;
    }
  }
  // GEMM2: gate[16x32] = h[16x128] x conv2_w^T
  v8f g0 = vzero8(), g1 = vzero8();
#pragma unroll
  for (int kk = 0; kk < 4; ++kk) {
    const int k0 = kk * 32;
    v16bf ah = frag_a(&Hs[w][0][0], REDc + 8, k0, lane);
    g0 = wmma_bf16(ah, frag_b(w2b, REDc, 0, k0, lane), g0);
    g1 = wmma_bf16(ah, frag_b(w2b, REDc, 16, k0, lane), g1);
  }
  const float cb0 = conv2_b[col], cb1 = conv2_b[16 + col];
#pragma unroll
  for (int v = 0; v < 8; ++v) {
    const int gr = row0 + v + 8 * half;
    gate[(size_t)gr * Ggrp + col] = g0[v] + cb0;
    gate[(size_t)gr * Ggrp + 16 + col] = g1[v] + cb1;
  }
}

// ------------- main fused kernel: out2 GEMM + involution out1 + LN + residuals
// block = 256 threads (8 waves), one 32-row tile; each wave owns 64 output
// cols x 32 rows (2 row-tiles x 4 col-tiles), K = 512, double-buffered.
__global__ void __launch_bounds__(256) k_out2_ln(
    const float* __restrict__ x, const __bf16* __restrict__ xb,
    const __bf16* __restrict__ Wb, const float* __restrict__ bbv,
    const float* __restrict__ gate, const float* __restrict__ ln_g,
    const float* __restrict__ ln_b, float* __restrict__ out) {
  __shared__ __align__(16) __bf16 As[32 * Dch];  // 32 KB A tile
  __shared__ float Ys[32][Dch + 4];              // out2 staging for LN
  const int tid = threadIdx.x;
  const int w = tid >> 5, lane = tid & 31;
  const int col = lane & 15, half = lane >> 4;
  const int row0 = blockIdx.x * 32;
  const int bat = row0 / Nlen;

  // ---- stage x tile (32 x 512 bf16) into LDS via the Tensor Data Mover ----
  const __bf16* gsrc = xb + (size_t)row0 * Dch;
#if HAVE_TDM
  if (w == 0) tdm_tile_512bf16(gsrc, (unsigned)(uintptr_t)(&As[0]), 32);
#else
  {
    const uint4* s4 = reinterpret_cast<const uint4*>(gsrc);
    uint4* d4 = reinterpret_cast<uint4*>(&As[0]);
    for (int i = tid; i < 32 * Dch / 8; i += 256) d4[i] = s4[i];
  }
#endif
  __syncthreads();

  // ---- out2 tile: [32 x 64] per wave ----
  const __bf16* Bmat = Wb + (size_t)bat * Dch * Dch;
  const int obase = w * 64;
  v8f acc[2][4];
#pragma unroll
  for (int m = 0; m < 2; ++m)
#pragma unroll
    for (int t = 0; t < 4; ++t) acc[m][t] = vzero8();

  v16bf afm[2];
  afm[0] = frag_a(As, Dch, 0, lane);
  afm[1] = frag_a(As + 16 * Dch, Dch, 0, lane);
  v16bf bf4[4];
#pragma unroll
  for (int t = 0; t < 4; ++t) bf4[t] = frag_b(Bmat, Dch, obase + t * 16, 0, lane);

#pragma unroll 1
  for (int kk = 0; kk < 16; ++kk) {
    const int kn = (kk < 15) ? (kk + 1) * 32 : 0;  // next-step fragments
    v16bf afn[2];
    afn[0] = frag_a(As, Dch, kn, lane);
    afn[1] = frag_a(As + 16 * Dch, Dch, kn, lane);
    v16bf bfn[4];
#pragma unroll
    for (int t = 0; t < 4; ++t) bfn[t] = frag_b(Bmat, Dch, obase + t * 16, kn, lane);
    __builtin_prefetch(Bmat + (size_t)(obase + col) * Dch + ((kn + 32) & (Dch - 1)), 0, 0);
#pragma unroll
    for (int t = 0; t < 4; ++t) {
      acc[0][t] = wmma_bf16(afm[0], bf4[t], acc[0][t]);
      acc[1][t] = wmma_bf16(afm[1], bf4[t], acc[1][t]);
    }
    afm[0] = afn[0]; afm[1] = afn[1];
#pragma unroll
    for (int t = 0; t < 4; ++t) bf4[t] = bfn[t];
  }
  // + dynamic bias, park in LDS for the row-wise LayerNorm
#pragma unroll
  for (int t = 0; t < 4; ++t) {
    const int o = obase + t * 16 + col;
    const float bo = bbv[bat * Dch + o];
#pragma unroll
    for (int m = 0; m < 2; ++m)
#pragma unroll
      for (int v = 0; v < 8; ++v) Ys[m * 16 + v + 8 * half][o] = acc[m][t][v] + bo;
  }
  __syncthreads();

  // ---- out1 + LN + double residual: wave w owns rows 4w..4w+3 ----
#pragma unroll 1
  for (int rr = 0; rr < 2; ++rr) {
    const int myrow = 4 * w + 2 * rr + half;
    const int gr = row0 + myrow;
    const float* xr = x + (size_t)gr * Dch;
    const float* grow = gate + (size_t)gr * Ggrp;
    float yv[32], xv[32];
    float s1 = 0.f, s2 = 0.f;
#pragma unroll
    for (int j = 0; j < 32; ++j) {
      const int d = j * 16 + col;  // channel group of d is exactly j
      const float xvv = xr[d];
      const float y = Ys[myrow][d] + xvv * grow[j];
      yv[j] = y; xv[j] = xvv;
      s1 += y; s2 += y * y;
    }
#pragma unroll
    for (int mm = 1; mm < 16; mm <<= 1) {  // 16-lane tree reduce (wave32)
      s1 += __shfl_xor(s1, mm, 32);
      s2 += __shfl_xor(s2, mm, 32);
    }
    const float mean = s1 * (1.f / Dch);
    const float var = s2 * (1.f / Dch) - mean * mean;
    const float rstd = rsqrtf(var + EPSc);
    float* orow = out + (size_t)gr * Dch;
#pragma unroll
    for (int j = 0; j < 32; ++j) {
      const int d = j * 16 + col;
      orow[d] = (yv[j] - mean) * rstd * ln_g[d] + ln_b[d] + xv[j];
    }
  }
}

// ---------------------------------------------------------------- launcher
extern "C" void kernel_launch(void* const* d_in, const int* in_sizes, int n_in,
                              void* d_out, int out_size, void* d_ws, size_t ws_size,
                              hipStream_t stream) {
  const float* x      = (const float*)d_in[0];
  const float* conv1w = (const float*)d_in[1];
  const float* bng    = (const float*)d_in[2];
  const float* bnb    = (const float*)d_in[3];
  const float* bnm    = (const float*)d_in[4];
  const float* bnv    = (const float*)d_in[5];
  const float* conv2w = (const float*)d_in[6];
  const float* conv2b = (const float*)d_in[7];
  const float* fc1w   = (const float*)d_in[8];
  const float* fc2w   = (const float*)d_in[9];
  const float* fc2b   = (const float*)d_in[10];
  const float* dynw   = (const float*)d_in[11];
  const float* dynb   = (const float*)d_in[12];
  const float* lng    = (const float*)d_in[13];
  const float* lnb    = (const float*)d_in[14];
  float* out = (float*)d_out;

  char* ws = (char*)d_ws;
  size_t off = 0;
  auto carve = [&](size_t bytes) {
    char* p = ws + off;
    off += (bytes + 255) & ~(size_t)255;
    return p;
  };
  const size_t NX = (size_t)Bdim * Nlen * Dch;
  __bf16* xb     = (__bf16*)carve(NX * 2);                         // 32 MiB
  __bf16* WbB    = (__bf16*)carve((size_t)Bdim * Dch * Dch * 2);   //  4 MiB
  __bf16* w1b    = (__bf16*)carve((size_t)REDc * Dch * 2);
  __bf16* w2b    = (__bf16*)carve((size_t)Ggrp * REDc * 2);
  float*  pooled = (float*)carve((size_t)Bdim * Dch * 4);
  float*  bbv    = (float*)carve((size_t)Bdim * Dch * 4);
  float*  gateb  = (float*)carve((size_t)Bdim * Nlen * Ggrp * 4);  //  4 MiB

  k_f32_to_bf16<<<(int)(NX / 256), 256, 0, stream>>>(x, xb, (int)NX);
  k_f32_to_bf16<<<(REDc * Dch) / 256, 256, 0, stream>>>(conv1w, w1b, REDc * Dch);
  k_f32_to_bf16<<<(Ggrp * REDc) / 256, 256, 0, stream>>>(conv2w, w2b, Ggrp * REDc);
  k_pool<<<Bdim, Dch, 0, stream>>>(x, pooled);
  k_att<<<Bdim, 256, 0, stream>>>(pooled, fc1w, fc2w, fc2b, dynw, dynb, WbB, bbv);
  k_gate<<<(Bdim * Nlen) / 64, 128, 0, stream>>>(xb, w1b, bng, bnb, bnm, bnv,
                                                 w2b, conv2b, gateb);
  k_out2_ln<<<(Bdim * Nlen) / 32, 256, 0, stream>>>(x, xb, WbB, bbv, gateb,
                                                    lng, lnb, out);
}
